// LILSTMCell_13864154432121
// MI455X (gfx1250) — matrix operand order repaired and checked by reading, not compiled, not run on hardware
//
#include <hip/hip_runtime.h>
#include <hip/hip_bf16.h>
#include <cstdint>

// ---------------- problem sizes (from reference) ----------------
#define T_STEPS 512
#define B_SZ    64
#define D_SZ    1024
#define H_SZ    1024
#define G4H     (4 * H_SZ)

// ---------------- vector types ----------------
typedef __attribute__((ext_vector_type(16))) __bf16 bf16x16;
typedef __attribute__((ext_vector_type(8)))  __bf16 bf16x8;
typedef __attribute__((ext_vector_type(8)))  float  f32x8;
typedef __attribute__((ext_vector_type(4)))  unsigned int u32x4;
typedef __attribute__((ext_vector_type(4)))  int    i32x4;
typedef __attribute__((ext_vector_type(8)))  int    i32x8;

union Frag16 { bf16x16 v; bf16x8 h[2]; };

#if defined(__AMDGCN__) && __has_builtin(__builtin_amdgcn_tensor_load_to_lds)
#define HAVE_TDM 1
#else
#define HAVE_TDM 0
#endif

// =================================================================
// Prep: convert Wi/Wh to bf16, fuse biases, init h (bf16) and c (f32)
// =================================================================
__global__ __launch_bounds__(256) void lstm_prep(
    const float* __restrict__ Wi, const float* __restrict__ Wh,
    const float* __restrict__ bi, const float* __restrict__ bh,
    const float* __restrict__ h0, const float* __restrict__ c0,
    __bf16* __restrict__ Wi_bf, __bf16* __restrict__ Wh_bf,
    float* __restrict__ bias, __bf16* __restrict__ h_bf,
    float* __restrict__ c_ws)
{
    const int stride = gridDim.x * blockDim.x;
    const int g      = blockIdx.x * blockDim.x + threadIdx.x;

    const int n_w = G4H * D_SZ;            // Wi and Wh have identical size (D==H)
    for (int i = g; i < n_w; i += stride) {
        Wi_bf[i] = (__bf16)Wi[i];
        Wh_bf[i] = (__bf16)Wh[i];
    }
    for (int i = g; i < G4H; i += stride)
        bias[i] = bi[i] + bh[i];
    for (int i = g; i < B_SZ * H_SZ; i += stride) {
        h_bf[i] = (__bf16)h0[i];
        c_ws[i] = c0[i];
    }
}

// =================================================================
// One LSTM time step.
//  grid = (H/16, B/16), block = 128 threads (4 waves).
//  Block (jt, mt) owns rows m0..m0+15 and h-columns j0..j0+15.
//  Wave w computes the 16x16 tile of gate w (pre columns w*H + j0 ...).
//  pre = x[t] @ Wi^T + h @ Wh^T + (bi + bh), all via bf16 WMMA, f32 acc.
//  h-staging goes through the Tensor Data Mover (async DMA -> LDS) when
//  the toolchain exposes the builtin.
// =================================================================
__global__ __launch_bounds__(128) void lstm_step(
    const float*  __restrict__ x_t,      // [B, D]  f32, this timestep
    const __bf16* __restrict__ Wi_bf,    // [4H, D] bf16
    const __bf16* __restrict__ Wh_bf,    // [4H, H] bf16
    const float*  __restrict__ bias,     // [4H]    f32 (bi + bh)
    const __bf16* __restrict__ h_in,     // [B, H]  bf16 (previous h)
    __bf16*       __restrict__ h_out_bf, // [B, H]  bf16 (next h)
    float*        __restrict__ c_ws,     // [B, H]  f32 cell state (in/out)
    float*        __restrict__ hs_t,     // [B, H]  f32 output slice hs[t]
    float*        __restrict__ h_fin,    // null or [B, H] final-h output
    float*        __restrict__ c_fin)    // null or [B, H] final-c output
{
    constexpr int KC   = 512;       // K staging chunk
    constexpr int LDAc = KC + 8;    // padded LDS row stride (bf16 elems) = 16B pad

    __shared__ __bf16 Abuf[16 * LDAc];      // staged A tile (x or h), bf16
    __shared__ float  gl[4][16][17];        // gate exchange buffer

    const int tid  = threadIdx.x;
    const int wave = tid >> 5;              // 0..3 -> gate f,i,o,g
    const int lane = tid & 31;
    const int m0   = blockIdx.y * 16;       // batch-row tile
    const int j0   = blockIdx.x * 16;       // h-column tile

    // --- accumulator initialized with fused bias (per output column) ---
    const float bv = bias[wave * H_SZ + j0 + (lane & 15)];
    f32x8 acc = {bv, bv, bv, bv, bv, bv, bv, bv};

    // --- two GEMM phases: phase 0 = x @ Wi^T, phase 1 = h @ Wh^T ---
    for (int phase = 0; phase < 2; ++phase) {
        const __bf16* __restrict__ W = (phase == 0) ? Wi_bf : Wh_bf;
        const int Kdim = (phase == 0) ? D_SZ : H_SZ;

        for (int kc = 0; kc < Kdim; kc += KC) {
            // ---- stage A chunk into LDS ----
            if (phase == 0) {
                // x is f32: load float4, convert to bf16 (whole block)
                for (int v = tid; v < 16 * (KC / 4); v += 128) {
                    const int r  = v >> 7;            // / (KC/4)
                    const int cc = (v & 127) << 2;
                    const float4 xv =
                        *(const float4*)(x_t + (size_t)(m0 + r) * D_SZ + kc + cc);
                    __bf16* dst = &Abuf[r * LDAc + cc];
                    dst[0] = (__bf16)xv.x; dst[1] = (__bf16)xv.y;
                    dst[2] = (__bf16)xv.z; dst[3] = (__bf16)xv.w;
                }
            } else {
#if HAVE_TDM
                // Tensor Data Mover: one DMA descriptor copies the 16x512
                // bf16 tile from global h into LDS, inserting a 16B pad
                // every 512 elements to reproduce the LDAc row stride.
                if (wave == 0) {
                    const uint64_t ga =
                        (uint64_t)(uintptr_t)(h_in + (size_t)m0 * H_SZ + kc);
                    const unsigned lds_base = (unsigned)(uintptr_t)(void*)&Abuf[0];

                    // D# group 0: count=1 | lds_addr | global_addr | type=2
                    u32x4 g0 = {
                        1u,
                        lds_base,
                        (unsigned)(ga & 0xFFFFFFFFu),
                        (unsigned)((ga >> 32) & 0x01FFFFFFu) | (2u << 30)
                    };
                    // D# group 1: data_size=2B, pad_enable,
                    // pad_interval=256 DWORDs (code 7), pad_amount=4 DWORDs (code 3)
                    // tensor_dim0=1024, tensor_dim1=64,
                    // tile_dim0=512, tile_dim1=16, tensor_dim0_stride=1024
                    i32x8 g1 = {
                        (int)((1u << 16) | (1u << 20) | (7u << 22) | (3u << 25)),
                        (int)((1024u & 0xFFFFu) << 16),
                        (int)(((1024u >> 16) & 0xFFFFu) | ((64u & 0xFFFFu) << 16)),
                        (int)(((64u >> 16) & 0xFFFFu) | (512u << 16)),
                        (int)(16u),
                        (int)(1024u),
                        0,
                        0
                    };
                    const i32x4 gz = {0, 0, 0, 0};
#if __clang_major__ >= 23
                    const i32x8 gz8 = {0, 0, 0, 0, 0, 0, 0, 0};
                    __builtin_amdgcn_tensor_load_to_lds(g0, g1, gz, gz, gz8, 0);
#else
                    __builtin_amdgcn_tensor_load_to_lds(g0, g1, gz, gz, 0);
#endif
                    __builtin_amdgcn_s_wait_tensorcnt(0);
                }
#else
                // fallback: cooperative 16B bf16 copies
                for (int v = tid; v < 16 * (KC / 8); v += 128) {
                    const int r  = v >> 6;            // / (KC/8)
                    const int cc = (v & 63) << 3;
                    *(bf16x8*)&Abuf[r * LDAc + cc] =
                        *(const bf16x8*)(h_in + (size_t)(m0 + r) * H_SZ + kc + cc);
                }
#endif
            }
            __syncthreads();

            // ---- K loop: 16 WMMA (16x16x32 bf16) per chunk per wave ----
            const int arow = (lane & 15) * LDAc;
            const int half = lane >> 4;
            #pragma unroll 4
            for (int it = 0; it < KC / 32; ++it) {
                const int klocal = it * 32;

                // A fragment from LDS: lane = (M, half); two 16B pieces
                Frag16 a;
                a.h[0] = *(const bf16x8*)&Abuf[arow + klocal + half * 8];
                a.h[1] = *(const bf16x8*)&Abuf[arow + klocal + 16 + half * 8];

                // B fragment from global (L2-resident weights):
                // row n = gate*H + j0 + (lane&15), two 16B pieces (K half*8, 16+half*8)
                const __bf16* brow =
                    W + (size_t)(wave * H_SZ + j0 + (lane & 15)) * Kdim
                      + kc + klocal;
                Frag16 b;
                b.h[0] = *(const bf16x8*)(brow + half * 8);
                b.h[1] = *(const bf16x8*)(brow + 16 + half * 8);

                if (it == 0 && kc + KC < Kdim)
                    __builtin_prefetch(brow + KC, 0, 1);   // global_prefetch

                acc = __builtin_amdgcn_wmma_f32_16x16x32_bf16(
                          /*neg_a=*/false, a.v, /*neg_b=*/false, b.v,
                          /*c_mod=*/(short)0, acc,
                          /*reuse_a=*/false, /*reuse_b=*/false);
            }
            __syncthreads();   // Abuf will be re-staged
        }
    }

    // --- nonlinearity per gate, publish tile to LDS ---
    // C layout: VGPR r, lane L -> (m = r + 8*(L>>4), n = L&15)
    {
        const int half = lane >> 4;
        const int n    = lane & 15;
        #pragma unroll
        for (int r = 0; r < 8; ++r) {
            const float p = acc[r];
            const float gv = (wave == 3) ? tanhf(p)
                                         : 1.0f / (1.0f + __expf(-p));
            gl[wave][r + 8 * half][n] = gv;
        }
    }
    __syncthreads();

    // --- combine gates: c' = f*c + i*g ; h = o*tanh(c') ---
    for (int e = tid; e < 256; e += 128) {
        const int m   = e >> 4;
        const int n   = e & 15;
        const int row = m0 + m;
        const int col = j0 + n;
        const size_t idx = (size_t)row * H_SZ + col;

        const float f = gl[0][m][n];
        const float i = gl[1][m][n];
        const float o = gl[2][m][n];
        const float g = gl[3][m][n];

        const float c_new = f * c_ws[idx] + i * g;
        c_ws[idx] = c_new;
        const float hv = o * tanhf(c_new);
        hs_t[idx]      = hv;
        h_out_bf[idx]  = (__bf16)hv;
        if (h_fin) { h_fin[idx] = hv; c_fin[idx] = c_new; }
    }
}

// =================================================================
// Host-side launcher
// =================================================================
extern "C" void kernel_launch(void* const* d_in, const int* in_sizes, int n_in,
                              void* d_out, int out_size, void* d_ws, size_t ws_size,
                              hipStream_t stream) {
    (void)in_sizes; (void)n_in; (void)out_size; (void)ws_size;

    const float* inputs = (const float*)d_in[0];   // (T, B, D)
    const float* h0     = (const float*)d_in[1];   // (1, B, H)
    const float* c0     = (const float*)d_in[2];   // (1, B, H)
    const float* Wi     = (const float*)d_in[3];   // (4H, D)
    const float* bi     = (const float*)d_in[4];   // (4H,)
    const float* Wh     = (const float*)d_in[5];   // (4H, H)
    const float* bh     = (const float*)d_in[6];   // (4H,)

    float* out   = (float*)d_out;
    float* hs    = out;                                        // T*B*H
    float* h_fin = out + (size_t)T_STEPS * B_SZ * H_SZ;        // B*H
    float* c_fin = h_fin + (size_t)B_SZ * H_SZ;                // B*H

    // ---- workspace layout (all offsets 256B aligned) ----
    char* ws = (char*)d_ws;
    const size_t W_BYTES = (size_t)G4H * D_SZ * sizeof(__bf16);   // 8 MB each
    __bf16* Wi_bf = (__bf16*)(ws);
    __bf16* Wh_bf = (__bf16*)(ws + W_BYTES);
    float*  bias  = (float*) (ws + 2 * W_BYTES);                  // 16 KB
    __bf16* hbuf0 = (__bf16*)(ws + 2 * W_BYTES + 65536);
    __bf16* hbuf1 = hbuf0 + (size_t)B_SZ * H_SZ;
    float*  c_ws  = (float*)((char*)(hbuf1 + (size_t)B_SZ * H_SZ));

    // 1) prep: weights->bf16, bias fuse, state init
    lstm_prep<<<1024, 256, 0, stream>>>(Wi, Wh, bi, bh, h0, c0,
                                        Wi_bf, Wh_bf, bias, hbuf0, c_ws);

    // 2) 512 sequential step kernels (graph-capture friendly)
    const dim3 grid(H_SZ / 16, B_SZ / 16);   // 64 x 4 blocks
    for (int t = 0; t < T_STEPS; ++t) {
        const __bf16* hin  = (t & 1) ? hbuf1 : hbuf0;
        __bf16*       hout = (t & 1) ? hbuf0 : hbuf1;
        const bool last = (t == T_STEPS - 1);
        lstm_step<<<grid, 128, 0, stream>>>(
            inputs + (size_t)t * B_SZ * D_SZ,
            Wi_bf, Wh_bf, bias, hin, hout, c_ws,
            hs + (size_t)t * B_SZ * H_SZ,
            last ? h_fin : nullptr,
            last ? c_fin : nullptr);
    }
}
